// FlowEstimatorRAFT_84963043049797
// MI455X (gfx1250) — matrix operand order repaired and checked by reading, not compiled, not run on hardware
//
#include <hip/hip_runtime.h>
#include <math.h>

// ---------------------------------------------------------------------------
// RAFT optical flow for MI455X (gfx1250, wave32, WMMA).
// Convs + correlation einsum run on v_wmma_f32_16x16x32_f16 via implicit
// GEMM. LDS staging uses CDNA5 async global->LDS copies
// (GLOBAL_LOAD_ASYNC_TO_LDS_B32, ASYNCcnt) so the DMA of chunk c+1 overlaps
// the WMMA of chunk c; out-of-bounds taps read a zeroed page instead of
// using divergent guards. Epilogues fuse bias + leaky-ReLU; GRU concats are
// consumed implicitly (3-source conv), never materialized.
// ---------------------------------------------------------------------------

typedef __attribute__((ext_vector_type(16))) _Float16 v16h;
typedef __attribute__((ext_vector_type(8)))  float    v8f;

#define LRELU(x) ((x) >= 0.f ? (x) : 0.1f * (x))

// CDNA5 async copy: per-lane global addr -> per-lane LDS addr, ASYNCcnt.
#define ASYNC_LDS_LOAD_B32(ldsaddr, gaddr)                                  \
    asm volatile("global_load_async_to_lds_b32 %0, %1, off"                 \
                 :: "v"(ldsaddr), "v"(gaddr) : "memory")
#define WAIT_ASYNC0() asm volatile("s_wait_asynccnt 0x0" ::: "memory")

static inline int ceil_div(int a, int b) { return (a + b - 1) / b; }

// ===========================================================================
// Implicit-GEMM convolution, NCHW, f16 WMMA, f32 accumulation.
// Input = implicit concat of up to 3 sources along channels (Ca,Cb,Cc).
// Block = 256 threads = 8 waves. Tile = 64 Cout x 32 pixels
// (waves arranged 4x2 over 16x16 WMMA tiles). K = Cin*KH*KW in chunks of 32,
// ping-pong double-buffered via async global->LDS DMA.
// grid = (ceil(P/32), ceil(Cout/64), N)
// ===========================================================================
__global__ __launch_bounds__(256)
void conv_wmma_k(const float* __restrict__ Xa, const float* __restrict__ Xb,
                 const float* __restrict__ Xc, int Ca, int Cb, int Cc,
                 const float* __restrict__ W, const float* __restrict__ Bias,
                 const float* __restrict__ zpage, float* __restrict__ Y,
                 int Hin, int Win, int Cout, int Hout, int Wout,
                 int KH, int KW, int stride, int pad, int act)
{
    // f32 staging (async copies move raw dwords); convert during frag build.
    __shared__ float smem[2 * 64 * 33 + 2 * 32 * 33];
    float* As32 = smem;                   // [2][64][33]
    float* Bs32 = smem + 2 * 64 * 33;     // [2][32][33]
    const unsigned ldsA0 = (unsigned)(size_t)(void*)As32;
    const unsigned ldsB0 = (unsigned)(size_t)(void*)Bs32;

    const int tid  = threadIdx.x;
    const int wave = tid >> 5;
    const int lane = tid & 31;
    const int col  = lane & 15;
    const int hi   = lane >> 4;

    const int P   = Hout * Wout;
    const int p0  = blockIdx.x * 32;
    const int co0 = blockIdx.y * 64;
    const int nb  = blockIdx.z;
    const int Cin = Ca + Cb + Cc;
    const int K   = Cin * KH * KW;
    const int KHW = KH * KW;

    const int mt = (wave & 3) * 16;   // M offset within 64
    const int nt = (wave >> 2) * 16;  // N offset within 32

    auto stageA = [&](int kk, int bi) {
        for (int idx = tid; idx < 64 * 32; idx += 256) {
            int m = idx >> 5, k = idx & 31;
            int co = co0 + m, kg = kk + k;
            const float* gp = (co < Cout && kg < K) ? &W[(size_t)co * K + kg]
                                                    : &zpage[lane];
            unsigned lds = ldsA0 + (unsigned)(((bi * 64 + m) * 33 + k) * 4);
            ASYNC_LDS_LOAD_B32(lds, (unsigned long long)(size_t)gp);
        }
    };
    auto stageB = [&](int kk, int bi) {
        for (int idx = tid; idx < 32 * 32; idx += 256) {
            int n = idx >> 5, k = idx & 31;
            int p = p0 + n, kg = kk + k;
            const float* gp = &zpage[lane];
            if (p < P && kg < K) {
                int ci = kg / KHW;
                int r  = kg - ci * KHW;
                int ky = r / KW, kx = r - ky * KW;
                int oy = p / Wout, ox = p - oy * Wout;
                int iy = oy * stride - pad + ky;
                int ix = ox * stride - pad + kx;
                if (iy >= 0 && iy < Hin && ix >= 0 && ix < Win) {
                    const float* src; int cl = ci, Cs;
                    if (cl < Ca)            { src = Xa; Cs = Ca; }
                    else if (cl < Ca + Cb)  { src = Xb; cl -= Ca; Cs = Cb; }
                    else                    { src = Xc; cl -= Ca + Cb; Cs = Cc; }
                    gp = &src[(((size_t)nb * Cs + cl) * Hin + iy) * Win + ix];
                }
            }
            unsigned lds = ldsB0 + (unsigned)(((bi * 32 + n) * 33 + k) * 4);
            ASYNC_LDS_LOAD_B32(lds, (unsigned long long)(size_t)gp);
        }
    };

    v8f acc = {};
    const int nch = (K + 31) >> 5;

    stageA(0, 0);
    stageB(0, 0);
    WAIT_ASYNC0();
    __syncthreads();

    for (int c = 0; c < nch; ++c) {
        const int cur = c & 1;
        if (c + 1 < nch) {                 // DMA next chunk while WMMA runs
            stageA((c + 1) * 32, cur ^ 1);
            stageB((c + 1) * 32, cur ^ 1);
        }
        if (c + 2 < nch && tid < 64) {     // warm L2/WGP$ for weight stream
            int co = co0 + tid;
            if (co < Cout)
                __builtin_prefetch(&W[(size_t)co * K + (c + 2) * 32], 0, 1);
        }

        // Fragments per CDNA5 16-bit A(16x32)/B(32x16) layouts (cvt f32->f16).
        const float* Ar = &As32[(cur * 64 + mt + col) * 33];
        const float* Br = &Bs32[(cur * 32 + nt + col) * 33];
        v16h a, b;
        #pragma unroll
        for (int vi = 0; vi < 8; ++vi) {
            int ka = (vi < 4 ? 2 * vi : 16 + 2 * (vi - 4)) + hi * 8;
            a[2 * vi]     = (_Float16)Ar[ka];
            a[2 * vi + 1] = (_Float16)Ar[ka + 1];
            int kb = (vi < 4 ? 2 * vi : 16 + 2 * (vi - 4)) + hi * 8;
            b[2 * vi]     = (_Float16)Br[kb];
            b[2 * vi + 1] = (_Float16)Br[kb + 1];
        }
        acc = __builtin_amdgcn_wmma_f32_16x16x32_f16(
                  false, a, false, b, (short)0, acc, false, false);
        WAIT_ASYNC0();                     // own async loads for next buffer
        __syncthreads();                   // publish buffer workgroup-wide
    }

    // C/D layout: VGPR r -> M = r + hi*8, N = col. Fused bias + activation.
    #pragma unroll
    for (int r = 0; r < 8; ++r) {
        int m = co0 + mt + r + hi * 8;
        int n = p0 + nt + col;
        if (m < Cout && n < P) {
            float o = acc[r] + Bias[m];
            if (act) o = LRELU(o);
            Y[((size_t)nb * Cout + m) * P + n] = o;
        }
    }
}

// ===========================================================================
// Correlation GEMM: C[bz][m][n] = scale * sum_k F1[bz][k][m] * F2[bz][k][n]
// F1/F2: (B, K, P). Output (B, P, P). P 64-aligned; K 32-aligned.
// Block = 256 threads = 8 waves, 64x64 tile; 2 accumulators per wave so the
// A fragment is reused. Async-DMA double-buffered LDS.
// grid = (P/64, P/64, B)
// ===========================================================================
__global__ __launch_bounds__(256)
void corr_wmma_k(const float* __restrict__ F1, const float* __restrict__ F2,
                 float* __restrict__ C, int K, int P, float scale)
{
    __shared__ float smem[2 * 64 * 33 * 2];
    float* As32 = smem;                   // [2][64][33]
    float* Bs32 = smem + 2 * 64 * 33;     // [2][64][33]
    const unsigned ldsA0 = (unsigned)(size_t)(void*)As32;
    const unsigned ldsB0 = (unsigned)(size_t)(void*)Bs32;

    const int tid  = threadIdx.x;
    const int wave = tid >> 5;
    const int lane = tid & 31;
    const int col  = lane & 15;
    const int hi   = lane >> 4;

    const int n0 = blockIdx.x * 64;
    const int m0 = blockIdx.y * 64;
    const int bz = blockIdx.z;
    const float* A  = F1 + (size_t)bz * K * P;
    const float* Bm = F2 + (size_t)bz * K * P;

    const int mt  = (wave & 3) * 16;
    const int nt0 = (wave >> 2) * 16;   // second tile at nt0+32

    auto stage = [&](int kk, int bi) {
        // rows of F1/F2 (coalesced), DMA'd transposed into [m][k] tiles
        for (int idx = tid; idx < 32 * 64; idx += 256) {
            int k = idx >> 6, m = idx & 63;
            unsigned e = (unsigned)(((bi * 64 + m) * 33 + k) * 4);
            ASYNC_LDS_LOAD_B32(ldsA0 + e,
                (unsigned long long)(size_t)&A[(size_t)(kk + k) * P + m0 + m]);
            ASYNC_LDS_LOAD_B32(ldsB0 + e,
                (unsigned long long)(size_t)&Bm[(size_t)(kk + k) * P + n0 + m]);
        }
    };

    v8f acc0 = {}, acc1 = {};
    const int nch = K >> 5;

    stage(0, 0);
    WAIT_ASYNC0();
    __syncthreads();

    for (int c = 0; c < nch; ++c) {
        const int cur = c & 1;
        if (c + 1 < nch) stage((c + 1) * 32, cur ^ 1);

        const float* Ar  = &As32[(cur * 64 + mt + col) * 33];
        const float* Br0 = &Bs32[(cur * 64 + nt0 + col) * 33];
        const float* Br1 = &Bs32[(cur * 64 + nt0 + 32 + col) * 33];
        v16h a, b0, b1;
        #pragma unroll
        for (int vi = 0; vi < 8; ++vi) {
            int ka = (vi < 4 ? 2 * vi : 16 + 2 * (vi - 4)) + hi * 8;
            a[2 * vi]     = (_Float16)Ar[ka];
            a[2 * vi + 1] = (_Float16)Ar[ka + 1];
            int kb = (vi < 4 ? 2 * vi : 16 + 2 * (vi - 4)) + hi * 8;
            b0[2 * vi]     = (_Float16)Br0[kb];
            b0[2 * vi + 1] = (_Float16)Br0[kb + 1];
            b1[2 * vi]     = (_Float16)Br1[kb];
            b1[2 * vi + 1] = (_Float16)Br1[kb + 1];
        }
        acc0 = __builtin_amdgcn_wmma_f32_16x16x32_f16(
                   false, a, false, b0, (short)0, acc0, false, false);
        acc1 = __builtin_amdgcn_wmma_f32_16x16x32_f16(
                   false, a, false, b1, (short)0, acc1, false, false);
        WAIT_ASYNC0();
        __syncthreads();
    }

    #pragma unroll
    for (int r = 0; r < 8; ++r) {
        int m = m0 + mt + r + hi * 8;
        size_t row = (size_t)bz * P * P + (size_t)m * P;
        C[row + n0 + nt0 + col]      = acc0[r] * scale;
        C[row + n0 + nt0 + 32 + col] = acc1[r] * scale;
    }
}

// ===========================================================================
// Instance norm (population var), optional fused leaky-ReLU; in-place.
// One block per (n,c) plane.
// ===========================================================================
__global__ __launch_bounds__(256)
void inorm_k(float* __restrict__ X, int HW, int lrelu_flag)
{
    __shared__ float sbuf[256], s2buf[256];
    float* p = X + (size_t)blockIdx.x * HW;
    float s = 0.f, s2 = 0.f;
    for (int i = threadIdx.x; i < HW; i += 256) {
        float v = p[i]; s += v; s2 += v * v;
    }
    sbuf[threadIdx.x] = s; s2buf[threadIdx.x] = s2;
    __syncthreads();
    for (int off = 128; off > 0; off >>= 1) {
        if ((int)threadIdx.x < off) {
            sbuf[threadIdx.x]  += sbuf[threadIdx.x + off];
            s2buf[threadIdx.x] += s2buf[threadIdx.x + off];
        }
        __syncthreads();
    }
    float mean = sbuf[0] / (float)HW;
    float var  = s2buf[0] / (float)HW - mean * mean;
    float inv  = rsqrtf(var + 1e-5f);
    for (int i = threadIdx.x; i < HW; i += 256) {
        float v = (p[i] - mean) * inv;
        if (lrelu_flag) v = LRELU(v);
        p[i] = v;
    }
}

// ======================= element-wise helpers ==============================
__global__ void add_lrelu_k(float* a, const float* __restrict__ b, long long n) {
    long long i = (long long)blockIdx.x * blockDim.x + threadIdx.x;
    if (i < n) { float v = a[i] + b[i]; a[i] = LRELU(v); }
}
__global__ void fill0_k(float* x, long long n) {
    long long i = (long long)blockIdx.x * blockDim.x + threadIdx.x;
    if (i < n) x[i] = 0.f;
}
__global__ void axpy_k(float* a, const float* __restrict__ b, long long n) {
    long long i = (long long)blockIdx.x * blockDim.x + threadIdx.x;
    if (i < n) a[i] += b[i];
}
// hidden = tanh(c[:, :128]); ctx = relu(c[:, 128:])
__global__ void split_act_k(const float* __restrict__ c, float* __restrict__ hid,
                            float* __restrict__ ctx, int B, int HW)
{
    long long i = (long long)blockIdx.x * blockDim.x + threadIdx.x;
    long long total = (long long)B * 256 * HW;
    if (i >= total) return;
    int s  = (int)(i % HW);
    int ch = (int)((i / HW) % 256);
    int b  = (int)(i / ((long long)HW * 256));
    float v = c[i];
    if (ch < 128) hid[((size_t)b * 128 + ch) * HW + s] = tanhf(v);
    else          ctx[((size_t)b * 128 + (ch - 128)) * HW + s] = fmaxf(v, 0.f);
}
// rh = sigmoid(rc) * h
__global__ void sigmul_k(const float* __restrict__ rc, const float* __restrict__ h,
                         float* __restrict__ rh, long long n)
{
    long long i = (long long)blockIdx.x * blockDim.x + threadIdx.x;
    if (i < n) { float z = 1.f / (1.f + expf(-rc[i])); rh[i] = z * h[i]; }
}
// h = (1 - sigmoid(zc))*h + sigmoid(zc)*tanh(qc)
__global__ void gru_up_k(float* __restrict__ h, const float* __restrict__ zc,
                         const float* __restrict__ qc, long long n)
{
    long long i = (long long)blockIdx.x * blockDim.x + threadIdx.x;
    if (i < n) {
        float z = 1.f / (1.f + expf(-zc[i]));
        h[i] = (1.f - z) * h[i] + z * tanhf(qc[i]);
    }
}
// 2x2 average pool over (NP, H, W) planes
__global__ void avgpool2_k(const float* __restrict__ in, float* __restrict__ out,
                           long long NP, int H, int W)
{
    int Ho = H >> 1, Wo = W >> 1;
    long long i = (long long)blockIdx.x * blockDim.x + threadIdx.x;
    long long total = NP * Ho * Wo;
    if (i >= total) return;
    int x = (int)(i % Wo);
    int y = (int)((i / Wo) % Ho);
    long long n = i / ((long long)Wo * Ho);
    const float* p = in + (n * H + 2 * y) * (size_t)W + 2 * x;
    out[i] = 0.25f * (p[0] + p[1] + p[W] + p[W + 1]);
}

// ===========================================================================
// Correlation lookup: cf[b, lvl*81 + j, y, x] = bilinear(pyr[lvl], window)
// pyramid level l: planes (B*H*W, H>>l, W>>l), zero outside, window r=4.
// ===========================================================================
__global__ void corr_lookup_k(const float* __restrict__ l0, const float* __restrict__ l1,
                              const float* __restrict__ l2, const float* __restrict__ l3,
                              const float* __restrict__ flow, float* __restrict__ cf,
                              int B, int H, int W)
{
    long long i = (long long)blockIdx.x * blockDim.x + threadIdx.x;
    long long total = (long long)B * H * W * 4 * 81;
    if (i >= total) return;
    int j   = (int)(i % 81);
    int lvl = (int)((i / 81) % 4);
    long long pix = i / (81 * 4);
    int x = (int)(pix % W);
    int y = (int)((pix / W) % H);
    int b = (int)(pix / ((long long)W * H));

    const float* img;
    if (lvl == 0) img = l0; else if (lvl == 1) img = l1;
    else if (lvl == 2) img = l2; else img = l3;
    int Hc = H >> lvl, Wc = W >> lvl;
    float s = (float)(1 << lvl);

    float fx = flow[(((size_t)b * 2 + 0) * H + y) * W + x];
    float fy = flow[(((size_t)b * 2 + 1) * H + y) * W + x];
    float dy = (float)(j / 9 - 4);
    float dx = (float)(j % 9 - 4);
    float tx = ((float)x + fx) / s + dx;
    float ty = ((float)y + fy) / s + dy;

    const float* plane = img + ((size_t)b * H * W + (size_t)y * W + x) * (size_t)(Hc * Wc);
    float x0f = floorf(tx), y0f = floorf(ty);
    float wx = tx - x0f, wy = ty - y0f;
    int x0 = (int)x0f, y0 = (int)y0f;
    int x1 = x0 + 1, y1 = y0 + 1;

    auto g = [&](int yy, int xx) -> float {
        if (yy < 0 || yy >= Hc || xx < 0 || xx >= Wc) return 0.f;
        return plane[yy * Wc + xx];
    };
    float v = g(y0, x0) * (1.f - wx) * (1.f - wy)
            + g(y0, x1) * wx * (1.f - wy)
            + g(y1, x0) * (1.f - wx) * wy
            + g(y1, x1) * wx * wy;

    cf[(((size_t)b * 324) + lvl * 81 + j) * (size_t)(H * W) + (size_t)y * W + x] = v;
}

// ===========================================================================
// Convex 8x upsampling: softmax(9) weighted 3x3 edge-padded flow patches.
// One thread per (b, y, x, sub) with sub in 0..63 -> (sy,sx).
// ===========================================================================
__global__ void convex_up8_k(const float* __restrict__ flow, const float* __restrict__ mk,
                             float* __restrict__ up, int B, int H, int W)
{
    long long i = (long long)blockIdx.x * blockDim.x + threadIdx.x;
    long long total = (long long)B * H * W * 64;
    if (i >= total) return;
    int s = (int)(i & 63);
    long long pix = i >> 6;
    int x = (int)(pix % W);
    int y = (int)((pix / W) % H);
    int b = (int)(pix / ((long long)W * H));

    float m[9];
    float mx = -1e30f;
    #pragma unroll
    for (int k = 0; k < 9; ++k) {
        m[k] = mk[(((size_t)b * 576 + s * 9 + k) * H + y) * W + x];
        mx = fmaxf(mx, m[k]);
    }
    float sum = 0.f;
    #pragma unroll
    for (int k = 0; k < 9; ++k) { m[k] = expf(m[k] - mx); sum += m[k]; }
    float inv = 8.f / sum;    // fold the *8.0 flow scaling in

    int sy = s >> 3, sx = s & 7;
    int W8 = 8 * W;
    #pragma unroll
    for (int ch = 0; ch < 2; ++ch) {
        float acc = 0.f;
        #pragma unroll
        for (int k = 0; k < 9; ++k) {
            int yy = y + k / 3 - 1; yy = yy < 0 ? 0 : (yy >= H ? H - 1 : yy);
            int xx = x + k % 3 - 1; xx = xx < 0 ? 0 : (xx >= W ? W - 1 : xx);
            acc += m[k] * flow[(((size_t)b * 2 + ch) * H + yy) * W + xx];
        }
        up[((size_t)(b * 2 + ch) * (8 * H) + (y * 8 + sy)) * (size_t)W8 + (x * 8 + sx)] = acc * inv;
    }
}

// ===========================================================================
// Host-side orchestration
// ===========================================================================
struct CP { const float* w; const float* b; };
struct EncP { CP c0; CP c1[5]; CP c2[5]; CP sc[5]; bool has_sc[5]; };

static const float* g_zpage;   // set once per kernel_launch before any conv

static void conv3(hipStream_t s, const float* xa, const float* xb, const float* xc,
                  int Ca, int Cb, int Cc, CP p, float* y,
                  int B, int Hin, int Win, int Cout, int K, int stride, int pad, int act)
{
    int Hout = (Hin + 2 * pad - K) / stride + 1;
    int Wout = (Win + 2 * pad - K) / stride + 1;
    dim3 g((unsigned)ceil_div(Hout * Wout, 32), (unsigned)ceil_div(Cout, 64), (unsigned)B);
    conv_wmma_k<<<g, 256, 0, s>>>(xa, xb, xc, Ca, Cb, Cc, p.w, p.b, g_zpage, y,
                                  Hin, Win, Cout, Hout, Wout, K, K, stride, pad, act);
}
static void conv(hipStream_t s, const float* x, CP p, float* y, int B, int Cin,
                 int Hin, int Win, int Cout, int K, int stride, int pad, int act)
{
    conv3(s, x, nullptr, nullptr, Cin, 0, 0, p, y, B, Hin, Win, Cout, K, stride, pad, act);
}
static void inorm(hipStream_t s, float* x, int planes, int HW, int lrelu) {
    inorm_k<<<(unsigned)planes, 256, 0, s>>>(x, HW, lrelu);
}

#define EW_GRID(n) dim3((unsigned)((n + 255) / 256)), dim3(256)

// Encoder: img (B,1,512,512) -> (B,256,64,64) copied into `out`.
static void run_encoder(hipStream_t s, const EncP& e, const float* img, float* out,
                        float* buf0, float* buf1, float* buf2, int B)
{
    float* buf[3] = { buf0, buf1, buf2 };
    // c0: 7x7 s2 p3, 1->64, 512 -> 256
    conv(s, img, e.c0, buf[0], B, 1, 512, 512, 64, 7, 2, 3, 0);
    inorm(s, buf[0], B * 64, 256 * 256, 1);

    int xi = 0, C = 64, H = 256;
    const int outs[5]    = { 64, 96, 96, 256, 256 };
    const int strides[5] = { 1, 2, 1, 2, 1 };
    for (int i = 0; i < 5; ++i) {
        int Cout = outs[i], st = strides[i], Hout = H / st;
        float* x  = buf[xi];
        float* t1 = buf[(xi + 1) % 3];
        float* t2 = buf[(xi + 2) % 3];
        long long n2 = (long long)B * Cout * Hout * Hout;

        conv(s, x, e.c1[i], t1, B, C, H, H, Cout, 3, st, 1, 0);
        inorm(s, t1, B * Cout, Hout * Hout, 1);
        conv(s, t1, e.c2[i], t2, B, Cout, Hout, Hout, Cout, 3, 1, 1, 0);
        inorm(s, t2, B * Cout, Hout * Hout, 0);
        if (e.has_sc[i]) {
            conv(s, x, e.sc[i], t1, B, C, H, H, Cout, 1, st, 0, 0);
            inorm(s, t1, B * Cout, Hout * Hout, 0);
            add_lrelu_k<<<EW_GRID(n2), 0, s>>>(t2, t1, n2);
        } else {
            add_lrelu_k<<<EW_GRID(n2), 0, s>>>(t2, x, n2);
        }
        xi = (xi + 2) % 3;
        C = Cout; H = Hout;
    }
    hipMemcpyAsync(out, buf[xi], (size_t)B * 256 * 64 * 64 * sizeof(float),
                   hipMemcpyDeviceToDevice, s);
}

extern "C" void kernel_launch(void* const* d_in, const int* in_sizes, int n_in,
                              void* d_out, int out_size, void* d_ws, size_t ws_size,
                              hipStream_t stream)
{
    (void)n_in; (void)out_size; (void)ws_size;
    const int B  = in_sizes[0] / (2 * 512 * 512);   // images: (B,2,512,512)
    const int H  = 64, W = 64, HW = H * W, P = HW;  // 1/8-res grid
    const float* images = (const float*)d_in[0];

    // ---- parameter unpacking: jax tree-flatten order (alphabetical keys) ---
    int pi = 1;
    auto pull = [&](void) -> CP {
        CP p; p.b = (const float*)d_in[pi++]; p.w = (const float*)d_in[pi++]; return p;
    };
    auto pull_enc = [&](void) -> EncP {
        EncP e;
        const bool sc[5] = { false, true, false, true, false };
        e.c0 = pull();
        for (int i = 0; i < 5; ++i) {
            e.c1[i] = pull(); e.c2[i] = pull();
            e.has_sc[i] = sc[i];
            if (sc[i]) e.sc[i] = pull();
        }
        return e;
    };
    EncP cnet = pull_enc();
    EncP fnet = pull_enc();
    CP cc1 = pull(), cc2 = pull(), fc1 = pull(), fc2 = pull(), fuse = pull();
    CP fh1 = pull(), fh2 = pull();
    CP wq  = pull(), wr  = pull(), wz = pull();
    CP mh1 = pull(), mh2 = pull();

    // ---- workspace bump allocator ----
    char* base = (char*)d_ws;
    size_t off = 0;
    auto alloc = [&](size_t elems) -> float* {
        float* p = (float*)(base + off);
        off += ((elems * sizeof(float) + 255) / 256) * 256;
        return p;
    };
    float* zpage = alloc(64);                        // zero page for OOB taps
    const size_t encN = (size_t)B * 64 * 256 * 256;  // largest encoder tensor
    float* eb0 = alloc(encN);
    float* eb1 = alloc(encN);
    float* eb2 = alloc(encN);
    float* f1   = alloc((size_t)B * 256 * P);
    float* f2   = alloc((size_t)B * 256 * P);
    float* cfeat= alloc((size_t)B * 256 * P);
    float* hid  = alloc((size_t)B * 128 * P);
    float* ctx  = alloc((size_t)B * 128 * P);
    float* pyr0 = alloc((size_t)B * P * P);
    float* pyr1 = alloc((size_t)B * P * P / 4);
    float* pyr2 = alloc((size_t)B * P * P / 16);
    float* pyr3 = alloc((size_t)B * P * P / 64);
    float* cf   = alloc((size_t)B * 324 * P);
    float* flow = alloc((size_t)B * 2 * P);
    float* df   = alloc((size_t)B * 2 * P);
    float* mk   = alloc((size_t)B * 576 * P);
    float* t192 = alloc((size_t)B * 192 * P);
    float* cfe  = alloc((size_t)B * 128 * P);
    float* t64a = alloc((size_t)B * 64 * P);
    float* t64b = alloc((size_t)B * 64 * P);
    float* mot  = alloc((size_t)B * 128 * P);
    float* zc   = alloc((size_t)B * 128 * P);
    float* rc   = alloc((size_t)B * 128 * P);
    float* qc   = alloc((size_t)B * 128 * P);
    float* rh   = alloc((size_t)B * 128 * P);
    float* th   = alloc((size_t)B * 128 * P);
    float* i1c  = alloc((size_t)B * 512 * 512);
    float* i2c  = alloc((size_t)B * 512 * 512);
    float* out  = (float*)d_out;

    g_zpage = zpage;
    fill0_k<<<EW_GRID(64LL), 0, stream>>>(zpage, 64);

    // images are (B,2,512,512); stage contiguous (B,1,512,512) copies so the
    // encoder sees batch-stride Cin=1 tensors.
    for (int b = 0; b < B; ++b) {
        hipMemcpyAsync(i1c + (size_t)b * 512 * 512,
                       images + ((size_t)b * 2 + 0) * 512 * 512,
                       (size_t)512 * 512 * sizeof(float), hipMemcpyDeviceToDevice, stream);
        hipMemcpyAsync(i2c + (size_t)b * 512 * 512,
                       images + ((size_t)b * 2 + 1) * 512 * 512,
                       (size_t)512 * 512 * sizeof(float), hipMemcpyDeviceToDevice, stream);
    }

    // ---- encoders ----
    run_encoder(stream, fnet, i1c, f1, eb0, eb1, eb2, B);
    run_encoder(stream, fnet, i2c, f2, eb0, eb1, eb2, B);
    run_encoder(stream, cnet, i1c, cfeat, eb0, eb1, eb2, B);
    {
        long long n = (long long)B * 256 * P;
        split_act_k<<<EW_GRID(n), 0, stream>>>(cfeat, hid, ctx, B, HW);
    }

    // ---- correlation volume + pyramid (WMMA GEMM) ----
    {
        dim3 g((unsigned)(P / 64), (unsigned)(P / 64), (unsigned)B);
        corr_wmma_k<<<g, 256, 0, stream>>>(f1, f2, pyr0, 256, P, 1.f / 16.f);
        long long NP = (long long)B * P;
        long long n1 = NP * 32 * 32, n2 = NP * 16 * 16, n3 = NP * 8 * 8;
        avgpool2_k<<<EW_GRID(n1), 0, stream>>>(pyr0, pyr1, NP, 64, 64);
        avgpool2_k<<<EW_GRID(n2), 0, stream>>>(pyr1, pyr2, NP, 32, 32);
        avgpool2_k<<<EW_GRID(n3), 0, stream>>>(pyr2, pyr3, NP, 16, 16);
    }

    // ---- iterative updates ----
    {
        long long nf = (long long)B * 2 * P;
        fill0_k<<<EW_GRID(nf), 0, stream>>>(flow, nf);
    }
    const long long n128 = (long long)B * 128 * P;
    for (int it = 0; it < 4; ++it) {
        long long nlk = (long long)B * HW * 4 * 81;
        corr_lookup_k<<<EW_GRID(nlk), 0, stream>>>(pyr0, pyr1, pyr2, pyr3, flow, cf, B, H, W);

        // motion encoder (bias+lrelu fused into conv epilogues)
        conv(stream, cf,   cc1, t192, B, 324, H, W, 192, 1, 1, 0, 1);
        conv(stream, t192, cc2, cfe,  B, 192, H, W, 128, 3, 1, 1, 1);
        conv(stream, flow, fc1, t64a, B,   2, H, W,  64, 7, 1, 3, 1);
        conv(stream, t64a, fc2, t64b, B,  64, H, W,  64, 3, 1, 1, 1);
        // fuse( concat[cfe(128) | t64b(64)] ) -> motion, implicit concat
        conv3(stream, cfe, t64b, nullptr, 128, 64, 0, fuse, mot,
              B, H, W, 128, 3, 1, 1, 1);

        // GRU: hx = [h | motion | ctx] consumed implicitly (no concat copies)
        conv3(stream, hid, mot, ctx, 128, 128, 128, wz, zc, B, H, W, 128, 3, 1, 1, 0);
        conv3(stream, hid, mot, ctx, 128, 128, 128, wr, rc, B, H, W, 128, 3, 1, 1, 0);
        sigmul_k<<<EW_GRID(n128), 0, stream>>>(rc, hid, rh, n128);
        conv3(stream, rh,  mot, ctx, 128, 128, 128, wq, qc, B, H, W, 128, 3, 1, 1, 0);
        gru_up_k<<<EW_GRID(n128), 0, stream>>>(hid, zc, qc, n128);

        // flow head + mask head
        conv(stream, hid, fh1, th, B, 128, H, W, 128, 3, 1, 1, 1);
        conv(stream, th,  fh2, df, B, 128, H, W,   2, 3, 1, 1, 0);
        conv(stream, hid, mh1, th, B, 128, H, W, 128, 3, 1, 1, 1);
        conv(stream, th,  mh2, mk, B, 128, H, W, 576, 1, 1, 0, 0);

        { long long nf = (long long)B * 2 * P;
          axpy_k<<<EW_GRID(nf), 0, stream>>>(flow, df, nf); }

        // convex upsample into output slot `it`
        float* up = out + (size_t)it * B * 2 * 512 * 512;
        long long nu = (long long)B * HW * 64;
        convex_up8_k<<<EW_GRID(nu), 0, stream>>>(flow, mk, up, B, H, W);
    }
}
